// GraILStyleModel_43928925504177
// MI455X (gfx1250) — matrix operand order repaired and checked by reading, not compiled, not run on hardware
//
#include <hip/hip_runtime.h>
#include <hip/hip_bf16.h>

#define DEVINL __device__ __forceinline__

typedef __attribute__((ext_vector_type(16))) __bf16 v16bf;
typedef __attribute__((ext_vector_type(8)))  float  v8f;

constexpr int NN = 50000;   // nodes
constexpr int NE = 800000;  // edges
constexpr int D  = 64;      // hidden dim
constexpr int NR = 16;      // relations
constexpr int NQ = 8192;    // queries
constexpr int NL = 2;       // layers
constexpr int ETPB = 8;     // 16-edge tiles swept per wave in edge kernel

// packed-fragment sizes (elements): one KxN weight matrix -> (N/16)*(K/32)*32 lanes*16
constexpr int PK64  = 4 * 2 * 32 * 16;   // K=64  -> 4096 elems
constexpr int PK192 = 4 * 6 * 32 * 16;   // K=192 -> 12288 elems

// ---------------- workspace layout (bytes, 256-aligned) ----------------
constexpr size_t al256(size_t x) { return (x + 255) & ~size_t(255); }
constexpr size_t WS_XBF   = 0;                                       // [N,64] bf16
constexpr size_t WS_AGG   = WS_XBF   + al256((size_t)NN * D * 2);    // [N,64] f32 (self-loop out)
constexpr size_t WS_MSG   = WS_AGG   + al256((size_t)NN * D * 4);    // [N,64] f32 (edge msg accum)
constexpr size_t WS_DEG   = WS_MSG   + al256((size_t)NN * D * 4);    // [N]    f32
constexpr size_t WS_ESRC  = WS_DEG   + al256((size_t)NN * 4);        // [E] int, rel-sorted
constexpr size_t WS_EDST  = WS_ESRC  + al256((size_t)NE * 4);        // [E] int, rel-sorted
constexpr size_t WS_RELW  = WS_EDST  + al256((size_t)NE * 4);        // [2*16] packed K=64 mats
constexpr size_t WS_SELFW = WS_RELW  + al256((size_t)NL * NR * PK64 * 2);
constexpr size_t WS_REMB  = WS_SELFW + al256((size_t)NL * PK64 * 2); // [16,64] bf16 row-major
constexpr size_t WS_SCW1  = WS_REMB  + al256((size_t)NR * D * 2);    // packed K=192 mat
constexpr size_t WS_H     = WS_SCW1  + al256((size_t)PK192 * 2);     // [Q,64] f32
constexpr size_t WS_CNT   = WS_H     + al256((size_t)NQ * D * 4);    // [16] u32
constexpr size_t WS_OFFS  = WS_CNT   + 256;                          // [17] u32
constexpr size_t WS_CUR   = WS_OFFS  + 256;                          // [16] u32

// ---------------- fragment helpers ----------------
union FragU {
  v16bf v;
  uint4          u4[2];
  unsigned int   ui[8];
  unsigned short us[16];
};

DEVINL unsigned short f32_to_bf16(float f) {
  unsigned int u = __builtin_bit_cast(unsigned int, f);
  u += 0x7FFFu + ((u >> 16) & 1u);   // round-to-nearest-even
  return (unsigned short)(u >> 16);
}

DEVINL v16bf zero_frag() {
  FragU f;
#pragma unroll
  for (int i = 0; i < 8; ++i) f.ui[i] = 0u;
  return f.v;
}

// A matrix 16x32 bf16 fragment from a contiguous bf16 row (row = M = lane&15).
// half 0: K in {k0+0..7, k0+16..23}; half 1: K in {k0+8..15, k0+24..31} (ISA 7.12.2)
// -> two contiguous 16B runs per lane -> two b128 loads.
DEVINL v16bf load_a_row(const unsigned short* rowptr, int k0, int half) {
  FragU f;
  f.u4[0] = *(const uint4*)(rowptr + k0 + half * 8);
  f.u4[1] = *(const uint4*)(rowptr + k0 + 16 + half * 8);
  return f.v;
}

// B fragment from prepacked buffer: frag (nt, kc) for this lane is 16 contiguous bf16.
DEVINL v16bf load_b_packed(const unsigned short* base, int nt, int kc, int KC, int lane) {
  const unsigned short* p = base + ((size_t)((nt * KC + kc) * 32 + lane) << 4);
  FragU f;
  f.u4[0] = *(const uint4*)p;
  f.u4[1] = *(const uint4*)(p + 8);
  return f.v;
}

DEVINL v8f wmma_bf16(v16bf a, v16bf b, v8f c) {
  return __builtin_amdgcn_wmma_f32_16x16x32_bf16(
      /*neg_a=*/false, a, /*neg_b=*/false, b,
      /*c_mod=*/(short)0, c, /*reuse_a=*/false, /*reuse_b=*/false);
}

// ---------------- kernels ----------------
__global__ void convert_bf16_kernel(const float* __restrict__ src,
                                    unsigned short* __restrict__ dst, int n) {
  int i = blockIdx.x * blockDim.x + threadIdx.x;
  if (i < n) dst[i] = f32_to_bf16(src[i]);
}

// Pack nmats row-major f32 weight matrices [K,64] into B-fragment order:
// packed[mat][((nt*KC + kc)*32 + lane)*16 + idx], idx pairs along K per ISA layout.
__global__ void pack_b_kernel(const float* __restrict__ W,
                              unsigned short* __restrict__ packed,
                              int K, int nmats) {
  const int KC = K / 32;
  const int perMat = 4 * KC * 32 * 16;
  int i = blockIdx.x * blockDim.x + threadIdx.x;
  if (i >= nmats * perMat) return;
  int mat  = i / perMat;
  int rem  = i - mat * perMat;
  int idx  = rem & 15;
  int lane = (rem >> 4) & 31;
  int frag = rem >> 9;           // nt*KC + kc
  int kc   = frag % KC;
  int nt   = frag / KC;
  int col  = nt * 16 + (lane & 15);
  int half = lane >> 4;
  int p    = idx >> 1;
  int k    = kc * 32 + (idx < 8 ? half * 8 + idx : 16 + half * 8 + (idx - 8)) + 0 * p;
  packed[i] = f32_to_bf16(W[(size_t)mat * K * 64 + (size_t)k * 64 + col]);
}

__global__ void input_proj_kernel(const float* __restrict__ nf,
                                  const float* __restrict__ inw,
                                  const float* __restrict__ inb,
                                  unsigned short* __restrict__ xbf) {
  int i = blockIdx.x * blockDim.x + threadIdx.x;
  if (i >= NN * D) return;
  int n = i >> 6, d = i & 63;
  float s = inb[d];
#pragma unroll
  for (int f = 0; f < 4; ++f) s += nf[n * 4 + f] * inw[f * D + d];
  xbf[i] = f32_to_bf16(s > 0.f ? s : 0.f);
}

__global__ void degree_kernel(const int* __restrict__ eidx, float* __restrict__ deg) {
  int e = blockIdx.x * blockDim.x + threadIdx.x;
  if (e >= NE) return;
  unsafeAtomicAdd(&deg[eidx[NE + e]], 1.0f);
}

__global__ void count_types_kernel(const int* __restrict__ etype, unsigned int* __restrict__ cnt) {
  int e = blockIdx.x * blockDim.x + threadIdx.x;
  if (e >= NE) return;
  atomicAdd(&cnt[etype[e]], 1u);
}

__global__ void scan_types_kernel(const unsigned int* __restrict__ cnt,
                                  unsigned int* __restrict__ offs,
                                  unsigned int* __restrict__ cur) {
  if (threadIdx.x != 0 || blockIdx.x != 0) return;
  unsigned int s = 0;
  for (int r = 0; r < NR; ++r) { offs[r] = s; cur[r] = s; s += cnt[r]; }
  offs[NR] = s;
}

__global__ void scatter_edges_kernel(const int* __restrict__ eidx,
                                     const int* __restrict__ etype,
                                     unsigned int* __restrict__ cur,
                                     int* __restrict__ esrc, int* __restrict__ edst) {
  int e = blockIdx.x * blockDim.x + threadIdx.x;
  if (e >= NE) return;
  unsigned int pos = atomicAdd(&cur[etype[e]], 1u);
  esrc[pos] = eidx[e];
  edst[pos] = eidx[NE + e];
}

// agg = x @ self_w + self_b ; block = 4 waves, wave w owns 16-col tile w
__global__ void self_gemm_kernel(const unsigned short* __restrict__ xbf,
                                 const unsigned short* __restrict__ wpk,
                                 const float* __restrict__ bias,
                                 float* __restrict__ out) {
  const int lane = threadIdx.x & 31;
  const int nt   = threadIdx.x >> 5;
  const int n0   = nt * 16;
  const int row0 = blockIdx.x * 16;
  const int half = lane >> 4, m = lane & 15;
  const unsigned short* rowptr = xbf + (size_t)(row0 + m) * D;
  v16bf a0 = load_a_row(rowptr, 0, half);
  v16bf a1 = load_a_row(rowptr, 32, half);
  v16bf b0 = load_b_packed(wpk, nt, 0, 2, lane);
  v16bf b1 = load_b_packed(wpk, nt, 1, 2, lane);
  v8f c = {};
  c = wmma_bf16(a0, b0, c);
  c = wmma_bf16(a1, b1, c);
  const int col = lane & 15;
  const float b = bias[n0 + col];
#pragma unroll
  for (int v = 0; v < 8; ++v) {
    const int row = row0 + half * 8 + v;
    out[(size_t)row * D + n0 + col] = c[v] + b;
  }
}

// per-relation gathered GEMM fused with atomic scatter-add into msgacc[dst].
// grid = (ceil(maxTiles/ETPB), 16 relations), block = 32 (one wave).
// The wave keeps the relation's 8 B-fragments resident in VGPRs and sweeps
// ETPB 16-edge tiles, so per tile only the A gather + 8 WMMAs + scatter remain.
__global__ void edge_msg_kernel(const unsigned short* __restrict__ xbf,
                                const unsigned short* __restrict__ relwpk,
                                const int* __restrict__ esrc,
                                const int* __restrict__ edst,
                                const unsigned int* __restrict__ offs,
                                float* __restrict__ msgacc) {
  const int r = blockIdx.y;
  const unsigned int beg = offs[r];
  const unsigned int cnt = offs[r + 1] - beg;
  const unsigned int tile0 = blockIdx.x * (ETPB * 16);
  if (tile0 >= cnt) return;

  const int lane = threadIdx.x;
  const int half = lane >> 4, m = lane & 15;
  const int col  = lane & 15;

  // relation matrix resident in registers (8 fragments = whole 64x64)
  const unsigned short* w = relwpk + (size_t)r * PK64;
  v16bf bfr[4][2];
#pragma unroll
  for (int nt = 0; nt < 4; ++nt) {
    bfr[nt][0] = load_b_packed(w, nt, 0, 2, lane);
    bfr[nt][1] = load_b_packed(w, nt, 1, 2, lane);
  }

  for (int t = 0; t < ETPB; ++t) {
    const unsigned int t0 = tile0 + t * 16;
    if (t0 >= cnt) break;

    // gather A rows (x_bf16[src]) for this 16-edge tile
    const bool mvalid = (t0 + m) < cnt;
    const int srow = mvalid ? esrc[beg + t0 + m] : 0;
    const unsigned short* rowptr = xbf + (size_t)srow * D;
    v16bf a0 = load_a_row(rowptr, 0, half);
    v16bf a1 = load_a_row(rowptr, 32, half);
    if (!mvalid) { a0 = zero_frag(); a1 = zero_frag(); }

    // dst rows for scatter, hoisted out of the col-tile loop
    int dstv[8];
    bool rvalid[8];
#pragma unroll
    for (int v = 0; v < 8; ++v) {
      const unsigned int row = t0 + half * 8 + v;
      rvalid[v] = row < cnt;
      dstv[v] = rvalid[v] ? edst[beg + row] : 0;
    }

#pragma unroll
    for (int nt = 0; nt < 4; ++nt) {
      v8f c = {};
      c = wmma_bf16(a0, bfr[nt][0], c);
      c = wmma_bf16(a1, bfr[nt][1], c);
#pragma unroll
      for (int v = 0; v < 8; ++v) {
        if (rvalid[v]) {
          unsafeAtomicAdd(&msgacc[(size_t)dstv[v] * D + nt * 16 + col], c[v]);
        }
      }
    }
  }
}

__global__ void update_kernel(const float* __restrict__ aggself,
                              const float* __restrict__ msgacc,
                              const float* __restrict__ deg,
                              unsigned short* __restrict__ xbf) {
  int i = blockIdx.x * blockDim.x + threadIdx.x;
  if (i >= NN * D) return;
  float d = deg[i >> 6];
  d = d < 1.f ? 1.f : d;
  float v = aggself[i] + msgacc[i] / d;
  xbf[i] = f32_to_bf16(v > 0.f ? v : 0.f);
}

// h = relu(concat(z_h, z_t, r) @ sc_w1 + sc_b1) ; block = 4 waves / 16-query tile
__global__ void score_gemm_kernel(const unsigned short* __restrict__ xbf,
                                  const unsigned short* __restrict__ relemb,
                                  const unsigned short* __restrict__ scw1pk,
                                  const float* __restrict__ scb1,
                                  const int* __restrict__ heads,
                                  const int* __restrict__ tails,
                                  const int* __restrict__ rels,
                                  float* __restrict__ h) {
  const int lane = threadIdx.x & 31;
  const int nt   = threadIdx.x >> 5;
  const int n0   = nt * 16;
  const int q0   = blockIdx.x * 16;
  const int half = lane >> 4, m = lane & 15;
  const int q = q0 + m;
  const unsigned short* segs[3] = {
      xbf    + (size_t)heads[q] * D,
      xbf    + (size_t)tails[q] * D,
      relemb + (size_t)rels[q]  * D };
  v8f c = {};
#pragma unroll
  for (int kc = 0; kc < 6; ++kc) {
    const int k0 = kc * 32;
    v16bf a = load_a_row(segs[k0 >> 6], k0 & 63, half);
    v16bf b = load_b_packed(scw1pk, nt, kc, 6, lane);
    c = wmma_bf16(a, b, c);
  }
  const int col = lane & 15;
  const float bias = scb1[n0 + col];
#pragma unroll
  for (int v = 0; v < 8; ++v) {
    const int row = q0 + half * 8 + v;
    const float val = c[v] + bias;
    h[(size_t)row * D + n0 + col] = val > 0.f ? val : 0.f;
  }
}

__global__ void score_out_kernel(const float* __restrict__ h,
                                 const float* __restrict__ w2,
                                 const float* __restrict__ b2,
                                 float* __restrict__ out) {
  int q = blockIdx.x * blockDim.x + threadIdx.x;
  if (q >= NQ) return;
  float s = b2[0];
#pragma unroll
  for (int d = 0; d < D; ++d) s += h[(size_t)q * D + d] * w2[d];
  out[q] = s;
}

// ---------------- launch ----------------
extern "C" void kernel_launch(void* const* d_in, const int* in_sizes, int n_in,
                              void* d_out, int out_size, void* d_ws, size_t ws_size,
                              hipStream_t stream) {
  const float* node_feat = (const float*)d_in[0];
  const int*   edge_idx  = (const int*)d_in[1];
  const int*   edge_type = (const int*)d_in[2];
  const int*   heads     = (const int*)d_in[3];
  const int*   rels      = (const int*)d_in[4];
  const int*   tails     = (const int*)d_in[5];
  const float* in_w      = (const float*)d_in[6];
  const float* in_b      = (const float*)d_in[7];
  const float* rel_w     = (const float*)d_in[8];
  const float* self_w    = (const float*)d_in[9];
  const float* self_b    = (const float*)d_in[10];
  const float* rel_emb   = (const float*)d_in[11];
  const float* sc_w1     = (const float*)d_in[12];
  const float* sc_b1     = (const float*)d_in[13];
  const float* sc_w2     = (const float*)d_in[14];
  const float* sc_b2     = (const float*)d_in[15];
  float* out = (float*)d_out;

  char* ws = (char*)d_ws;
  unsigned short* xbf    = (unsigned short*)(ws + WS_XBF);
  float*          agg    = (float*)(ws + WS_AGG);
  float*          msg    = (float*)(ws + WS_MSG);
  float*          deg    = (float*)(ws + WS_DEG);
  int*            esrc   = (int*)(ws + WS_ESRC);
  int*            edst   = (int*)(ws + WS_EDST);
  unsigned short* relwp  = (unsigned short*)(ws + WS_RELW);
  unsigned short* selfwp = (unsigned short*)(ws + WS_SELFW);
  unsigned short* rembb  = (unsigned short*)(ws + WS_REMB);
  unsigned short* scw1p  = (unsigned short*)(ws + WS_SCW1);
  float*          hbuf   = (float*)(ws + WS_H);
  unsigned int*   cnt    = (unsigned int*)(ws + WS_CNT);
  unsigned int*   offs   = (unsigned int*)(ws + WS_OFFS);
  unsigned int*   cur    = (unsigned int*)(ws + WS_CUR);

  hipMemsetAsync(deg, 0, (size_t)NN * 4, stream);
  hipMemsetAsync(cnt, 0, NR * 4, stream);

  // weight prep: pack GEMM operands into B-fragment order; rel_emb stays row-major (A side)
  pack_b_kernel<<<(NL*NR*PK64 + 255) / 256, 256, 0, stream>>>(rel_w,  relwp,  64, NL * NR);
  pack_b_kernel<<<(NL*PK64    + 255) / 256, 256, 0, stream>>>(self_w, selfwp, 64, NL);
  pack_b_kernel<<<(PK192      + 255) / 256, 256, 0, stream>>>(sc_w1,  scw1p,  192, 1);
  convert_bf16_kernel<<<(NR*D + 255) / 256, 256, 0, stream>>>(rel_emb, rembb, NR * D);

  input_proj_kernel<<<(NN * D + 255) / 256, 256, 0, stream>>>(node_feat, in_w, in_b, xbf);

  const int egrid = (NE + 255) / 256;
  degree_kernel<<<egrid, 256, 0, stream>>>(edge_idx, deg);
  count_types_kernel<<<egrid, 256, 0, stream>>>(edge_type, cnt);
  scan_types_kernel<<<1, 32, 0, stream>>>(cnt, offs, cur);
  scatter_edges_kernel<<<egrid, 256, 0, stream>>>(edge_idx, edge_type, cur, esrc, edst);

  const int maxTiles  = (NE + 15) / 16;                 // worst case: all edges one relation
  const int edgeGridX = (maxTiles + ETPB - 1) / ETPB;   // tiles swept per wave
  for (int l = 0; l < NL; ++l) {
    hipMemsetAsync(msg, 0, (size_t)NN * D * 4, stream);
    self_gemm_kernel<<<NN / 16, 128, 0, stream>>>(
        xbf, selfwp + (size_t)l * PK64, self_b + (size_t)l * D, agg);
    edge_msg_kernel<<<dim3(edgeGridX, NR), 32, 0, stream>>>(
        xbf, relwp + (size_t)l * NR * PK64, esrc, edst, offs, msg);
    update_kernel<<<(NN * D + 255) / 256, 256, 0, stream>>>(agg, msg, deg, xbf);
  }

  score_gemm_kernel<<<NQ / 16, 128, 0, stream>>>(
      xbf, rembb, scw1p, sc_b1, heads, tails, rels, hbuf);
  score_out_kernel<<<(NQ + 255) / 256, 256, 0, stream>>>(hbuf, sc_w2, sc_b2, out);
}
